// LoRAMoE_25185688223877
// MI455X (gfx1250) — matrix operand order repaired and checked by reading, not compiled, not run on hardware
//
#include <hip/hip_runtime.h>

// ---------------------------------------------------------------------------
// LoRA-MoE fused layer for MI455X (gfx1250, wave32, WMMA bf16 16x16x32,
// async global->LDS staging with double buffering)
//
// S=2048 B=8 D_IN=1024 D_OUT=4096 E=8 R=16  -> M = S*B = 16384, ER = 128
// ---------------------------------------------------------------------------

typedef __bf16 bf16;
typedef __attribute__((ext_vector_type(16))) __bf16 bf16x16;
typedef __attribute__((ext_vector_type(8)))  __bf16 bf16x8;
typedef __attribute__((ext_vector_type(8)))  float  f32x8;
typedef int v4i __attribute__((vector_size(16)));   // matches builtin's V4i

#define AS1 __attribute__((address_space(1)))
#define AS3 __attribute__((address_space(3)))

#if __has_builtin(__builtin_amdgcn_global_load_async_to_lds_b128) && \
    __has_builtin(__builtin_amdgcn_s_wait_asynccnt)
#define USE_ASYNC_LDS 1
#else
#define USE_ASYNC_LDS 0
#endif

static constexpr int S_    = 2048;
static constexpr int B_    = 8;
static constexpr int DIN   = 1024;
static constexpr int DOUT  = 4096;
static constexpr int E_    = 8;
static constexpr int R_    = 16;
static constexpr int M_    = S_ * B_;   // 16384
static constexpr int ER    = E_ * R_;   // 128
static constexpr int LDS_PITCH = 72;    // 64 + 8 bf16 pad (16B) -> conflict-free b128 reads
static constexpr int TILE_ELEMS = 128 * LDS_PITCH;

__device__ __forceinline__ bf16x16 cat8(bf16x8 lo, bf16x8 hi) {
  return __builtin_shufflevector(lo, hi, 0,1,2,3,4,5,6,7,8,9,10,11,12,13,14,15);
}

// Issue one 128x64 bf16 tile: global (row-major, leading dim ld) -> LDS slab.
// 256 threads x 4 chunks of 16B each.
__device__ __forceinline__ void stage_issue(const bf16* __restrict__ g, bf16* s,
                                            int row_base, int ld, int k0) {
  #pragma unroll
  for (int i = 0; i < 4; ++i) {
    int cid = threadIdx.x + i * 256;
    int row = cid >> 3, c8 = (cid & 7) * 8;
    const bf16* gp = &g[(size_t)(row_base + row) * ld + k0 + c8];
    bf16*       lp = &s[row * LDS_PITCH + c8];
#if USE_ASYNC_LDS
    __builtin_amdgcn_global_load_async_to_lds_b128((AS1 v4i*)((void*)gp),
                                                   (AS3 v4i*)((void*)lp), 0, 0);
#else
    *(bf16x8*)lp = *(const bf16x8*)gp;
#endif
  }
}

__device__ __forceinline__ void stage_wait() {
#if USE_ASYNC_LDS
  __builtin_amdgcn_s_wait_asynccnt(0);
#endif
}

// One 64-wide K slab of WMMAs: wave tile 64x32 -> 4x2 accumulators.
__device__ __forceinline__ void mma_slab(const bf16* sA, const bf16* sB,
                                         f32x8 acc[4][2],
                                         int m_off, int n_off, int l15, int h) {
  #pragma unroll
  for (int kk = 0; kk < 64; kk += 32) {
    bf16x16 af[4], bfr[2];
    #pragma unroll
    for (int mt = 0; mt < 4; ++mt) {
      // A frag: lane l15 = row M, halves at K = 8h..8h+7 and 16+8h..23+8h
      const bf16* r = &sA[(m_off + mt * 16 + l15) * LDS_PITCH + kk + 8 * h];
      af[mt] = cat8(*(const bf16x8*)r, *(const bf16x8*)(r + 16));
    }
    #pragma unroll
    for (int nt = 0; nt < 2; ++nt) {
      // B frag: lane l15 = col N, halves at K = 8h..8h+7 and 16+8h..23+8h
      const bf16* r = &sB[(n_off + nt * 16 + l15) * LDS_PITCH + kk + 8 * h];
      bfr[nt] = cat8(*(const bf16x8*)r, *(const bf16x8*)(r + 16));
    }
    #pragma unroll
    for (int mt = 0; mt < 4; ++mt)
      #pragma unroll
      for (int nt = 0; nt < 2; ++nt)
        acc[mt][nt] = __builtin_amdgcn_wmma_f32_16x16x32_bf16(
            false, af[mt], false, bfr[nt], (short)0, acc[mt][nt], false, false);
  }
}

// ---------------- gate path ------------------------------------------------

__global__ void k_zero_f32(float* __restrict__ p, int n) {
  int i = blockIdx.x * 256 + threadIdx.x;
  if (i < n) p[i] = 0.0f;
}

// xsum[b*DIN + i] = sum_s x[(s*B+b)*DIN + i];  grid (32, 16) x 256, split-S atomics
__global__ void k_colsum(const float* __restrict__ x, float* __restrict__ xsum) {
  int idx = blockIdx.x * 256 + threadIdx.x;          // 0..8191 == b*1024 + i
  int s0  = blockIdx.y * 128;
  const float* p = x + (size_t)s0 * (B_ * DIN) + idx;
  float acc = 0.0f;
  #pragma unroll 8
  for (int s = 0; s < 128; ++s) acc += p[(size_t)s * (B_ * DIN)];
  atomicAdd(&xsum[idx], acc);
}

// g[b*E + e] = dot(xsum[b,:], gate_w[e,:]) / S + gate_b[e];  one wave per (b,e)
__global__ void k_gate(const float* __restrict__ xsum, const float* __restrict__ gw,
                       const float* __restrict__ gb, float* __restrict__ g) {
  int wid  = (blockIdx.x * 256 + threadIdx.x) >> 5;  // 64 waves
  int lane = threadIdx.x & 31;
  int b = wid >> 3, e = wid & 7;
  const float* xs = xsum + b * DIN;
  const float* w  = gw   + e * DIN;
  float acc = 0.0f;
  for (int i = lane; i < DIN; i += 32) acc += xs[i] * w[i];
  #pragma unroll
  for (int off = 16; off > 0; off >>= 1) acc += __shfl_xor(acc, off, 32);
  if (lane == 0) g[b * E_ + e] = acc * (1.0f / (float)S_) + gb[e];
}

// ---------------- operand preparation --------------------------------------

__global__ void k_cvt_f32_bf16(const float* __restrict__ in, bf16* __restrict__ out, int n) {
  int i = blockIdx.x * 256 + threadIdx.x;
  if (i < n) out[i] = (bf16)in[i];
}

// Bflat[o*ER + (e*R+r)] = lora_B[e, o, r]
__global__ void k_build_bflat(const float* __restrict__ lB, bf16* __restrict__ Bf) {
  int idx = blockIdx.x * 256 + threadIdx.x;          // DOUT*ER = 524288
  if (idx >= DOUT * ER) return;
  int o = idx >> 7, c = idx & 127, e = c >> 4, r = c & 15;
  Bf[idx] = (bf16)lB[((size_t)e * DOUT + o) * R_ + r];
}

// ---------------- WMMA GEMM 1: u = (x @ A^T) * gate  -----------------------
// M=16384, N=128(=ER), K=1024; block tile 128x128; 8 waves of 64x32.

__global__ __launch_bounds__(256) void k_lora_a(const bf16* __restrict__ xb,
                                                const bf16* __restrict__ Ab,
                                                const float* __restrict__ g,
                                                bf16* __restrict__ ub) {
  __shared__ __align__(16) bf16 sA[2][TILE_ELEMS];
  __shared__ __align__(16) bf16 sB[2][TILE_ELEMS];
  const int m_base = blockIdx.x * 128;
  const int lane = threadIdx.x & 31, wave = threadIdx.x >> 5;
  const int m_off = (wave >> 2) * 64, n_off = (wave & 3) * 32;
  const int l15 = lane & 15, h = lane >> 4;
  const int NSLAB = DIN / 64;                        // 16

  f32x8 acc[4][2] = {};

  stage_issue(xb, sA[0], m_base, DIN, 0);
  stage_issue(Ab, sB[0], 0,      DIN, 0);
  for (int t = 0; t < NSLAB; ++t) {
    const int cur = t & 1;
    stage_wait();
    __syncthreads();
    if (t + 1 < NSLAB) {
      stage_issue(xb, sA[cur ^ 1], m_base, DIN, (t + 1) * 64);
      stage_issue(Ab, sB[cur ^ 1], 0,      DIN, (t + 1) * 64);
    }
    mma_slab(sA[cur], sB[cur], acc, m_off, n_off, l15, h);
  }

  // epilogue: scale by gate weight, store bf16
  #pragma unroll
  for (int nt = 0; nt < 2; ++nt) {
    int col = n_off + nt * 16 + l15;   // c = e*R + r
    int e = col >> 4;
    #pragma unroll
    for (int mt = 0; mt < 4; ++mt) {
      #pragma unroll
      for (int j = 0; j < 8; ++j) {
        int m = m_base + m_off + mt * 16 + 8 * h + j;
        float sc = g[(m & 7) * E_ + e];          // b = m % B
        ub[(size_t)m * ER + col] = (bf16)(acc[mt][nt][j] * sc);
      }
    }
  }
}

// ---------------- WMMA GEMM 2: out = x@W^T + u@Bflat^T + bias --------------
// M=16384, N=4096; slabs 0..15: K=1024 over (x, W); slabs 16..17: K=128 over
// (u, Bflat). Double-buffered straight through the phase boundary.

__global__ __launch_bounds__(256) void k_main_gemm(const bf16* __restrict__ xb,
                                                   const bf16* __restrict__ Wb,
                                                   const bf16* __restrict__ ub,
                                                   const bf16* __restrict__ Bf,
                                                   const float* __restrict__ bias,
                                                   float* __restrict__ out) {
  __shared__ __align__(16) bf16 sA[2][TILE_ELEMS];
  __shared__ __align__(16) bf16 sB[2][TILE_ELEMS];
  const int m_base = blockIdx.x * 128;
  const int n_base = blockIdx.y * 128;
  const int lane = threadIdx.x & 31, wave = threadIdx.x >> 5;
  const int m_off = (wave >> 2) * 64, n_off = (wave & 3) * 32;
  const int l15 = lane & 15, h = lane >> 4;
  const int NSLAB = DIN / 64 + ER / 64;              // 16 + 2 = 18

  f32x8 acc[4][2] = {};

  auto issue_slab = [&](int t, int buf) {
    const bool base_phase = (t < DIN / 64);
    const bf16* gA = base_phase ? xb : ub;
    const bf16* gB = base_phase ? Wb : Bf;
    const int ld   = base_phase ? DIN : ER;
    const int k0   = base_phase ? t * 64 : (t - DIN / 64) * 64;
    stage_issue(gA, sA[buf], m_base, ld, k0);
    stage_issue(gB, sB[buf], n_base, ld, k0);
  };

  issue_slab(0, 0);
  for (int t = 0; t < NSLAB; ++t) {
    const int cur = t & 1;
    stage_wait();
    __syncthreads();
    if (t + 1 < NSLAB) issue_slab(t + 1, cur ^ 1);
    mma_slab(sA[cur], sB[cur], acc, m_off, n_off, l15, h);
  }

  // epilogue: + bias, store f32
  #pragma unroll
  for (int nt = 0; nt < 2; ++nt) {
    int o = n_base + n_off + nt * 16 + l15;
    float bv = bias[o];
    #pragma unroll
    for (int mt = 0; mt < 4; ++mt) {
      #pragma unroll
      for (int j = 0; j < 8; ++j) {
        int m = m_base + m_off + mt * 16 + 8 * h + j;
        out[(size_t)m * DOUT + o] = acc[mt][nt][j] + bv;
      }
    }
  }
}

// ---------------------------------------------------------------------------

extern "C" void kernel_launch(void* const* d_in, const int* in_sizes, int n_in,
                              void* d_out, int out_size, void* d_ws, size_t ws_size,
                              hipStream_t stream) {
  const float* x      = (const float*)d_in[0];
  const float* gate_w = (const float*)d_in[1];
  const float* gate_b = (const float*)d_in[2];
  const float* W      = (const float*)d_in[3];
  const float* bias   = (const float*)d_in[4];
  const float* lora_A = (const float*)d_in[5];
  const float* lora_B = (const float*)d_in[6];
  float* out = (float*)d_out;

  // workspace layout (bytes)
  char* ws = (char*)d_ws;
  bf16*  xb   = (bf16*)(ws);                      // M*DIN*2      = 33,554,432
  bf16*  Wb   = (bf16*)(ws + 33554432);           // DOUT*DIN*2   =  8,388,608
  bf16*  Ab   = (bf16*)(ws + 41943040);           // ER*DIN*2     =    262,144
  bf16*  Bf   = (bf16*)(ws + 42205184);           // DOUT*ER*2    =  1,048,576
  bf16*  ub   = (bf16*)(ws + 43253760);           // M*ER*2       =  4,194,304
  float* xsum = (float*)(ws + 47448064);          // B*DIN*4      =     32,768
  float* gbuf = (float*)(ws + 47480832);          // B*E*4        =        256

  // gate path
  k_zero_f32<<<32, 256, 0, stream>>>(xsum, B_ * DIN);
  k_colsum<<<dim3(32, 16), 256, 0, stream>>>(x, xsum);
  k_gate<<<8, 256, 0, stream>>>(xsum, gate_w, gate_b, gbuf);

  // bf16 operand prep
  k_cvt_f32_bf16<<<(M_ * DIN) / 256, 256, 0, stream>>>(x, xb, M_ * DIN);
  k_cvt_f32_bf16<<<(DOUT * DIN) / 256, 256, 0, stream>>>(W, Wb, DOUT * DIN);
  k_cvt_f32_bf16<<<(ER * DIN) / 256, 256, 0, stream>>>(lora_A, Ab, ER * DIN);
  k_build_bflat<<<(DOUT * ER) / 256, 256, 0, stream>>>(lora_B, Bf);

  // u = (x @ A^T) * g
  k_lora_a<<<M_ / 128, 256, 0, stream>>>(xb, Ab, gbuf, ub);

  // out = x @ W^T + u @ Bflat^T + bias
  k_main_gemm<<<dim3(M_ / 128, DOUT / 128), 256, 0, stream>>>(xb, Wb, ub, Bf, bias, out);
}
